// SentimentClassification_60078002537025
// MI455X (gfx1250) — matrix operand order repaired and checked by reading, not compiled, not run on hardware
//
#include <hip/hip_runtime.h>
#include <stdint.h>

typedef __attribute__((ext_vector_type(16))) __bf16 v16bf;
typedef __attribute__((ext_vector_type(2)))  __bf16 v2bf;
typedef __attribute__((ext_vector_type(8)))  float  v8f;
typedef int v4i_ __attribute__((vector_size(16)));

#define B_ 64
#define T_ 2048
#define E_ 256
#define H_ 256

// ---------------------------------------------------------------- bf16 helpers
__device__ __forceinline__ unsigned short bfbits(float f) {
    return __builtin_bit_cast(unsigned short, (__bf16)f);
}
__device__ __forceinline__ unsigned int pack_bf16(float x, float y) {
    v2bf p;
    p[0] = (__bf16)x;
    p[1] = (__bf16)y;
    return __builtin_bit_cast(unsigned int, p);
}
__device__ __forceinline__ float bf16_bits_to_f32(unsigned short s) {
    unsigned int u = ((unsigned int)s) << 16;
    return __builtin_bit_cast(float, u);
}

// ------------------------------------------------- async global->LDS (gfx1250)
#if defined(__has_builtin)
#if __has_builtin(__builtin_amdgcn_global_load_async_to_lds_b128)
#define HAVE_ASYNC_LDS 1
#endif
#endif
#ifndef HAVE_ASYNC_LDS
#define HAVE_ASYNC_LDS 0
#endif

__device__ __forceinline__ void async_b128(void* lds, const void* gsrc) {
#if HAVE_ASYNC_LDS
    __builtin_amdgcn_global_load_async_to_lds_b128(
        (__attribute__((address_space(1))) v4i_*)gsrc,
        (__attribute__((address_space(3))) v4i_*)lds,
        0, 0);
#else
    *(uint4*)lds = *(const uint4*)gsrc;   // synchronous fallback
#endif
}
__device__ __forceinline__ void async_wait_all() {
#if HAVE_ASYNC_LDS
#if __has_builtin(__builtin_amdgcn_s_wait_asynccnt)
    __builtin_amdgcn_s_wait_asynccnt(0);
#else
    asm volatile("s_wait_asynccnt 0x0" ::: "memory");
#endif
#endif
}

// -----------------------------------------------------------------------------
// Kernel 0: one-shot f32 -> bf16 conversion of W_ih and W_hh into workspace.
// Removes per-workgroup conversion VALU from the hot kernels and lets them
// stage weights with async b128 copies.
// -----------------------------------------------------------------------------
__global__ void __launch_bounds__(256) cvt_weights_kernel(
    const float* __restrict__ Wih, const float* __restrict__ Whh,
    unsigned int* __restrict__ WihP, unsigned int* __restrict__ WhhP)
{
    const int i = blockIdx.x * 256 + threadIdx.x;   // pair index, 32768 total
    float2 a = ((const float2*)Wih)[i];
    WihP[i] = pack_bf16(a.x, a.y);
    float2 b = ((const float2*)Whh)[i];
    WhhP[i] = pack_bf16(b.x, b.y);
}

// -----------------------------------------------------------------------------
// Kernel 1: fused embedding-gather + input-projection GEMM.
//   x_proj[t][b][h] = sum_e emb[tok[b][t]][e] * W_ih[h][e]  (bf16 out, [T,B,H])
// W_ih (bf16) staged into 128KB LDS via GLOBAL_LOAD_ASYNC_TO_LDS_B128.
// Each wave: 16-row M slab x full N=256 (16 accum tiles), K=256 in 8 chunks of
// 32 via v_wmma_f32_16x16x32_bf16.
// -----------------------------------------------------------------------------
__global__ void __launch_bounds__(256) xproj_kernel(
    const int*   __restrict__ reviews,        // [B,T] flattened, m = b*T + t
    const float* __restrict__ emb,            // [VOCAB, E] f32
    const unsigned short* __restrict__ Wihbf, // [H, E] bf16 (pre-converted)
    unsigned short* __restrict__ xp)          // [T, B, H] bf16
{
    extern __shared__ char smem[];
    unsigned short* sW = (unsigned short*)smem;   // [H][E] bf16 = 128KB

    const int tid = threadIdx.x;

    // Async-stage W_ih(bf16) -> LDS: 131072 B = 8192 x b128, 32 per thread.
    for (int c = tid; c < (H_ * E_) / 8; c += 256)
        async_b128(sW + c * 8, Wihbf + c * 8);
    async_wait_all();
    __syncthreads();

    const int lane = tid & 31;
    const int wave = tid >> 5;
    const int col  = lane & 15;       // N-column / A-row this lane serves
    const int hi   = lane >> 4;       // lane half selects K sub-range
    const int mBase = blockIdx.x * 128 + wave * 16;

    // Each lane gathers the embedding row for A-matrix row (lane & 15).
    const int  mrow = mBase + col;
    const int  tok  = reviews[mrow];
    const float* arow = emb + (long)tok * E_;

    v8f acc[16] = {};

    for (int kc = 0; kc < 8; ++kc) {
        // A fragment (16-bit A 16x32 layout): lane<16 -> K ks..ks+7 & ks+16..+23
        const int ks = kc * 32 + (hi ? 8 : 0);
        v16bf a;
#pragma unroll
        for (int j = 0; j < 8; ++j)
            a[j] = (__bf16)arow[ks + j];
#pragma unroll
        for (int j = 0; j < 8; ++j)
            a[8 + j] = (__bf16)arow[ks + 16 + j];

        const int kb = kc * 32 + (hi ? 8 : 0);
#pragma unroll
        for (int nt = 0; nt < 16; ++nt) {
            const unsigned short* pb = sW + (nt * 16 + col) * E_ + kb;
            union { v16bf v; uint4 q[2]; } bu;
            bu.q[0] = *(const uint4*)(pb);
            bu.q[1] = *(const uint4*)(pb + 16);
            acc[nt] = __builtin_amdgcn_wmma_f32_16x16x32_bf16(
                false, a, false, bu.v, (short)0, acc[nt], false, false);
        }
    }

    // Store: C/D layout -> element (M = r + 8*hi, N = col) of each tile.
#pragma unroll
    for (int r = 0; r < 8; ++r) {
        const int mg = mBase + r + 8 * hi;
        const int b  = mg >> 11;          // / T_
        const int t  = mg & (T_ - 1);     // % T_
        unsigned short* dst = xp + ((long)(t * B_ + b) * H_);
#pragma unroll
        for (int nt = 0; nt < 16; ++nt)
            dst[nt * 16 + col] = bfbits(acc[nt][r]);
    }
}

// -----------------------------------------------------------------------------
// Kernel 2: sequential RNN scan + last-state select + classifier.
// 4 workgroups, each owns 16 independent batch rows. W_hh bf16 async-staged in
// LDS; per-step xp slab (8KB) double-buffered in LDS with async loads issued
// one step ahead (latency of the global read is hidden inside the dependent
// t-chain). h kept as bf16 in LDS.
// -----------------------------------------------------------------------------
__global__ void __launch_bounds__(256) rnn_scan_kernel(
    const unsigned short* __restrict__ xp,    // [T,B,H] bf16
    const unsigned short* __restrict__ Whhbf, // [H,H] bf16 (pre-converted)
    const int*   __restrict__ lengths,        // [B]
    const float* __restrict__ Wcls,           // [2,H]
    const float* __restrict__ bcls,           // [2]
    float* __restrict__ out)                  // [B,2]
{
    extern __shared__ char smem[];
    unsigned short* sW    = (unsigned short*)smem;                 // 128KB  [256][256]
    unsigned short* sHbf  = (unsigned short*)(smem + 131072);      // 8KB    [16][256]
    float*          sLast = (float*)(smem + 139264);               // 16KB   [16][256]
    int*            sLen  = (int*)(smem + 155648);                 // 64B
    unsigned short* sXp   = (unsigned short*)(smem + 155712);      // 16KB   [2][16][256]

    const int tid  = threadIdx.x;
    const int lane = tid & 31;
    const int wave = tid >> 5;
    const int col  = lane & 15;
    const int hi   = lane >> 4;
    const int bbase = blockIdx.x * 16;

    // Async-stage W_hh(bf16) -> LDS.
    for (int c = tid; c < (H_ * H_) / 8; c += 256)
        async_b128(sW + c * 8, Whhbf + c * 8);
    for (int i = tid; i < (16 * H_) / 2; i += 256)
        ((unsigned int*)sHbf)[i] = 0;              // h0 = 0
    if (tid < 16) sLen[tid] = lengths[bbase + tid];

    // Prime double buffer with xp slab for t = 0 (8KB = 512 x b128, 2/thread).
    {
        const unsigned short* g = xp + (long)bbase * H_;
#pragma unroll
        for (int i = 0; i < 2; ++i) {
            const int c = tid + i * 256;
            async_b128(sXp + c * 8, g + c * 8);
        }
    }
    async_wait_all();
    __syncthreads();

    const int nt0 = wave * 2, nt1 = wave * 2 + 1;  // two N-tiles per wave

    for (int t = 0; t < T_; ++t) {
        const unsigned short* cur = sXp + (t & 1) * (16 * H_);

        // Issue async loads for step t+1 into the other buffer.
        if (t + 1 < T_) {
            const unsigned short* g = xp + (long)(t + 1) * (B_ * H_) + (long)bbase * H_;
            unsigned short* nxt = sXp + ((t + 1) & 1) * (16 * H_);
#pragma unroll
            for (int i = 0; i < 2; ++i) {
                const int c = tid + i * 256;
                async_b128(nxt + c * 8, g + c * 8);
            }
        }

        v8f acc0 = {}, acc1 = {};
#pragma unroll
        for (int r = 0; r < 8; ++r) {
            const int m = r + 8 * hi;
            acc0[r] = bf16_bits_to_f32(cur[m * H_ + nt0 * 16 + col]);
            acc1[r] = bf16_bits_to_f32(cur[m * H_ + nt1 * 16 + col]);
        }

#pragma unroll
        for (int kc = 0; kc < 8; ++kc) {
            const int ks = kc * 32 + (hi ? 8 : 0);
            union { v16bf v; uint4 q[2]; } au;
            au.q[0] = *(const uint4*)(sHbf + col * H_ + ks);
            au.q[1] = *(const uint4*)(sHbf + col * H_ + ks + 16);

            const int kb = kc * 32 + (hi ? 8 : 0);
            union { v16bf v; uint4 q[2]; } b0, b1;
            const unsigned short* pb0 = sW + (nt0 * 16 + col) * H_ + kb;
            const unsigned short* pb1 = sW + (nt1 * 16 + col) * H_ + kb;
            b0.q[0] = *(const uint4*)pb0; b0.q[1] = *(const uint4*)(pb0 + 16);
            b1.q[0] = *(const uint4*)pb1; b1.q[1] = *(const uint4*)(pb1 + 16);

            acc0 = __builtin_amdgcn_wmma_f32_16x16x32_bf16(
                false, au.v, false, b0.v, (short)0, acc0, false, false);
            acc1 = __builtin_amdgcn_wmma_f32_16x16x32_bf16(
                false, au.v, false, b1.v, (short)0, acc1, false, false);
        }

        __syncthreads();   // all waves done reading sHbf for this step

#pragma unroll
        for (int r = 0; r < 8; ++r) {
            const int m = r + 8 * hi;
            const float v0 = tanhf(acc0[r]);
            const float v1 = tanhf(acc1[r]);
            sHbf[m * H_ + nt0 * 16 + col] = bfbits(v0);
            sHbf[m * H_ + nt1 * 16 + col] = bfbits(v1);
            if (sLen[m] == t + 1) {        // hs[b, len-1] snapshot
                sLast[m * H_ + nt0 * 16 + col] = v0;
                sLast[m * H_ + nt1 * 16 + col] = v1;
            }
        }

        async_wait_all();  // xp slab for t+1 resident before next iteration
        __syncthreads();   // h(t) + slab visible to all waves
    }

    // Classifier: 32 lanes -> (row, class) pairs.
    if (tid < 32) {
        const int row = tid & 15;
        const int cls = tid >> 4;
        float s = bcls[cls];
        for (int h = 0; h < H_; ++h)
            s += sLast[row * H_ + h] * Wcls[cls * H_ + h];
        out[(bbase + row) * 2 + cls] = s;
    }
}

// -----------------------------------------------------------------------------
extern "C" void kernel_launch(void* const* d_in, const int* in_sizes, int n_in,
                              void* d_out, int out_size, void* d_ws, size_t ws_size,
                              hipStream_t stream) {
    const int*   reviews = (const int*)d_in[0];
    const int*   lengths = (const int*)d_in[1];
    const float* emb     = (const float*)d_in[2];
    const float* Wih     = (const float*)d_in[3];
    const float* Whh     = (const float*)d_in[4];
    const float* Wcls    = (const float*)d_in[5];
    const float* bcls    = (const float*)d_in[6];
    float* out = (float*)d_out;

    // Workspace layout:
    //   [0, 64MB)           x_proj bf16 [T,B,H]
    //   [64MB, +128KB)      W_ih bf16
    //   [64MB+128KB, +128K) W_hh bf16
    char* ws = (char*)d_ws;
    unsigned short* xp    = (unsigned short*)ws;
    unsigned int*   WihP  = (unsigned int*)(ws + (size_t)67108864);
    unsigned int*   WhhP  = (unsigned int*)(ws + (size_t)67108864 + 131072);

    cvt_weights_kernel<<<dim3(128), dim3(256), 0, stream>>>(Wih, Whh, WihP, WhhP);

    // 131072 rows / 128 rows-per-WG = 1024 workgroups; 128KB dynamic LDS each.
    xproj_kernel<<<dim3(1024), dim3(256), 131072, stream>>>(
        reviews, emb, (const unsigned short*)WihP, xp);

    // 4 workgroups x 16 batch rows; LDS = 128K + 8K + 16K + 64B + 16K = 172096.
    rnn_scan_kernel<<<dim3(4), dim3(256), 172096, stream>>>(
        xp, (const unsigned short*)WhhP, lengths, Wcls, bcls, out);
}
